// AverageBackProjection_14130442404022
// MI455X (gfx1250) — matrix operand order repaired and checked by reading, not compiled, not run on hardware
//
#include <hip/hip_runtime.h>

// ---------------------------------------------------------------------------
// Types
// ---------------------------------------------------------------------------
typedef __bf16 bf16_t;
typedef __attribute__((ext_vector_type(16))) __bf16 v16bf;
typedef __attribute__((ext_vector_type(8)))  __bf16 v8bf;
typedef __attribute__((ext_vector_type(8)))  float  v8f;

constexpr int N_VOX = 300000;
constexpr int GK    = 27;          // 3^3 kernel offsets
constexpr int TILES = N_VOX / 16;  // 18750, exact

static __device__ __forceinline__ bf16_t f2bf(float x) {
  // fp32 -> bf16, round-to-nearest-even
  unsigned int u = __float_as_uint(x);
  unsigned int r = u + 0x7FFFu + ((u >> 16) & 1u);
  unsigned short h = (unsigned short)(r >> 16);
  return __builtin_bit_cast(bf16_t, h);
}

// ---------------------------------------------------------------------------
// Weight repack: W[k][ci][co] fp32 -> bf16 WMMA B-fragments.
// Fragment f = (k*CC + cc)*CJ + j covers K-rows [cc*32, cc*32+32) and
// columns [j*16, j*16+16).  Contiguous-K 16-bit B 32x16 layout used here (wave32):
//   lane L: n = L&15 (column), kh = L>>4 (K half 0..15 / 16..31)
//   vgpr r holds B[kh*16 + 2r, n] (lo16) and B[kh*16 + 2r + 1, n] (hi16)
// Stored flat: frag base f*512 elems, lane chunk = lane*16 elems (32 B).
// ---------------------------------------------------------------------------
__global__ void repack_kernel(const float* __restrict__ W, bf16_t* __restrict__ WR,
                              int CIN, int COUT) {
  const int CC = CIN >> 5, CJ = COUT >> 4;
  const int tid = blockIdx.x * blockDim.x + threadIdx.x;
  const int nf = GK * CC * CJ;
  const int f = tid >> 5;
  if (f >= nf) return;
  const int lane = tid & 31;
  const int j  = f % CJ;
  const int cc = (f / CJ) % CC;
  const int k  = f / (CJ * CC);
  const int n  = lane & 15, kh = lane >> 4;

  bf16_t* dst = WR + (size_t)f * 512 + lane * 16;
  const float* src = W + (size_t)k * CIN * COUT
                       + (size_t)(cc * 32 + kh * 16) * COUT + j * 16 + n;
#pragma unroll
  for (int r = 0; r < 8; ++r) {
    dst[2 * r]     = f2bf(src[(2 * r)     * COUT]);
    dst[2 * r + 1] = f2bf(src[(2 * r + 1) * COUT]);
  }
}

// ---------------------------------------------------------------------------
// Prep: downsample-mean + residual, fp32 -> bf16 into X0[(N+1) x 128].
// One wave per row; lane owns channels {lane, lane+32, lane+64, lane+96} so
// the group-mean is computed entirely in-lane (coalesced loads).
// Row N of X0/X1/X2 is zeroed (missing-neighbor gather target).
// ---------------------------------------------------------------------------
__global__ void prep_kernel(const float* __restrict__ feats,
                            bf16_t* __restrict__ X0,
                            bf16_t* __restrict__ X1,
                            bf16_t* __restrict__ X2) {
  const int wave = blockIdx.x * (blockDim.x >> 5) + (threadIdx.x >> 5);
  const int lane = threadIdx.x & 31;
  if (wave > N_VOX) return;
  if (wave == N_VOX) {
    bf16_t z = f2bf(0.f);
#pragma unroll
    for (int g = 0; g < 4; ++g) {
      X0[(size_t)N_VOX * 128 + lane + 32 * g] = z;
      X1[(size_t)N_VOX * 128 + lane + 32 * g] = z;
    }
    X2[(size_t)N_VOX * 32 + lane] = z;
    return;
  }
  const float* row = feats + (size_t)wave * 128;
  float x0 = row[lane], x1 = row[lane + 32], x2 = row[lane + 64], x3 = row[lane + 96];
  float m = 0.25f * (x0 + x1 + x2 + x3);
  bf16_t* o = X0 + (size_t)wave * 128;
  o[lane]      = f2bf(x0 - m);
  o[lane + 32] = f2bf(x1 - m);
  o[lane + 64] = f2bf(x2 - m);
  o[lane + 96] = f2bf(x3 - m);
}

// ---------------------------------------------------------------------------
// Sparse conv: Y[n] = sum_k X[nbr[k,n]] @ W[k], bf16 inputs, fp32 accumulate.
// Block = 256 threads = 8 waves; each wave owns MT voxel tiles of 16.
// The hot loop is branch-free: out-of-range tiles are clamped to tile 0
// (harmless redundant compute in the last partial block only) and the
// `act` predicate — derived from a readfirstlane'd scalar so it lowers to
// s_cmp/s_cbranch, not EXEC manipulation — guards only the epilogue stores.
// Pipelining:
//   - LDS weight slices double-buffered (2 x LDSB); global loads for slice
//     k+1 issue before the WMMA chain of slice k, ds_stores after it,
//     one barrier per k.
//   - neighbor indices prefetched one k ahead.
//   - A fragments double-buffered across cc chunks.
//   - B fragments in a 4-slot rotating file (prefetch distance 2, no copies)
//     so ds_load(j+2) issues before wmma(j); each B feeds MT WMMAs.
// FINAL: fuse out = downsampled(feats) + Y (fp32); else write bf16 to Ynext.
// ---------------------------------------------------------------------------
template <int CIN, int COUT, int MT, bool FINAL>
__global__ void __launch_bounds__(256)
conv_kernel(const bf16_t* __restrict__ X, const bf16_t* __restrict__ WR,
            const int* __restrict__ nbr, bf16_t* __restrict__ Ynext,
            float* __restrict__ Yfin, const float* __restrict__ feats) {
  constexpr int CC    = CIN / 32;        // K-chunks of 32
  constexpr int CJ    = COUT / 16;       // 16-col output tiles
  constexpr int LDSB  = CC * CJ * 1024;  // bytes per k-slice of repacked W
  constexpr int N16   = LDSB / 16;       // uint4 count per slice
  constexpr int PER_T = (N16 + 255) / 256;
  __shared__ unsigned char smem[2 * LDSB];

  const int lane = threadIdx.x & 31;
  // tid>>5 is identical across a wave; readfirstlane makes it provably scalar
  const int waveU = __builtin_amdgcn_readfirstlane(threadIdx.x >> 5);
  const int tile0 = (blockIdx.x * 8 + waveU) * MT;
  const int n  = lane & 15;
  const int kh = lane >> 4;

  bool act[MT];
  int tileC[MT];
  const int* nptr[MT];
#pragma unroll
  for (int m = 0; m < MT; ++m) {
    const int t = tile0 + m;
    act[m]   = (t < TILES);              // scalar condition
    tileC[m] = act[m] ? t : 0;           // clamp: compute is always valid
    nptr[m]  = nbr + tileC[m] * 16 + n;
  }

  auto loadA = [&](const bf16_t* row, int cc) -> v16bf {
    // Contiguous-K 16x32 A layout used here (same k order as B): lanes 0-15 take K[0..15] of the chunk,
    // lanes 16-31 take K[16..31].
    const bf16_t* ab = row + cc * 32 + kh * 16;
    v8bf alo = *(const v8bf*)(ab);
    v8bf ahi = *(const v8bf*)(ab + 8);
    v16bf a;
#pragma unroll
    for (int t = 0; t < 8; ++t) { a[t] = alo[t]; a[t + 8] = ahi[t]; }
    return a;
  };
  auto loadB = [&](const unsigned char* bbase, int cc, int j) -> v16bf {
    return *(const v16bf*)(bbase + (cc * CJ + j) * 1024 + lane * 32);
  };

  v8f acc[MT][CJ] = {};

  const uint4* __restrict__ wsrc = (const uint4*)WR;

  // ---- prologue: stage slice 0 into LDS buffer 0 ----
  {
    uint4 stg[PER_T];
#pragma unroll
    for (int t = 0; t < PER_T; ++t) {
      int i = threadIdx.x + t * 256;
      if (i < N16) stg[t] = wsrc[i];
    }
    uint4* dst = (uint4*)smem;
#pragma unroll
    for (int t = 0; t < PER_T; ++t) {
      int i = threadIdx.x + t * 256;
      if (i < N16) dst[i] = stg[t];
    }
  }
  __syncthreads();

  int idx[MT];
#pragma unroll
  for (int m = 0; m < MT; ++m) idx[m] = nptr[m][0];

  for (int k = 0; k < GK; ++k) {
    const bool more = (k + 1 < GK);

    // issue global loads for slice k+1 now; latency hides behind WMMA chain
    uint4 stg[PER_T];
    if (more) {
      const uint4* src = wsrc + (size_t)(k + 1) * N16;
#pragma unroll
      for (int t = 0; t < PER_T; ++t) {
        int i = threadIdx.x + t * 256;
        if (i < N16) stg[t] = src[i];
      }
    }
    int idxNext[MT];
#pragma unroll
    for (int m = 0; m < MT; ++m)
      idxNext[m] = more ? nptr[m][(size_t)(k + 1) * N_VOX] : idx[m];

    {
      const unsigned char* bbase = smem + (k & 1) * LDSB;
      const bf16_t* row[MT];
      v16bf a0[MT];
#pragma unroll
      for (int m = 0; m < MT; ++m) {
        row[m] = X + (size_t)idx[m] * CIN;
        a0[m] = loadA(row[m], 0);
      }
#pragma unroll
      for (int cc = 0; cc < CC; ++cc) {
        v16bf a1[MT];
#pragma unroll
        for (int m = 0; m < MT; ++m)
          a1[m] = (cc + 1 < CC) ? loadA(row[m], cc + 1) : a0[m];

        // 4-slot B pipeline, prefetch distance 2, no rotation copies
        v16bf b[4];
        b[0] = loadB(bbase, cc, 0);
        if (CJ > 1) b[1] = loadB(bbase, cc, 1);
#pragma unroll
        for (int j = 0; j < CJ; ++j) {
          if (j + 2 < CJ) b[(j + 2) & 3] = loadB(bbase, cc, j + 2);
#pragma unroll
          for (int m = 0; m < MT; ++m) {
            acc[m][j] = __builtin_amdgcn_wmma_f32_16x16x32_bf16(
                false, a0[m], false, b[j & 3], (short)0, acc[m][j], false, false);
          }
        }
#pragma unroll
        for (int m = 0; m < MT; ++m) a0[m] = a1[m];
      }
    }

    // commit slice k+1 into the alternate LDS buffer
    if (more) {
      uint4* dst = (uint4*)(smem + ((k + 1) & 1) * LDSB);
#pragma unroll
      for (int t = 0; t < PER_T; ++t) {
        int i = threadIdx.x + t * 256;
        if (i < N16) dst[i] = stg[t];
      }
    }
    __syncthreads();
#pragma unroll
    for (int m = 0; m < MT; ++m) idx[m] = idxNext[m];
  }

  // C/D layout: lane L -> column n=L&15; vgpr r -> row m = r + 8*(L>>4)
#pragma unroll
  for (int m = 0; m < MT; ++m) {
    if (!act[m]) continue;               // scalar branch, stores only
    const int voxBase = (tile0 + m) * 16;
#pragma unroll
    for (int j = 0; j < CJ; ++j) {
#pragma unroll
      for (int r = 0; r < 8; ++r) {
        const int mm = voxBase + 8 * kh + r;
        const int c = j * 16 + n;
        const float v = acc[m][j][r];
        if constexpr (FINAL) {
          const float* fr = feats + (size_t)mm * 128;
          const float dn = 0.25f * (fr[c] + fr[c + 32] + fr[c + 64] + fr[c + 96]);
          Yfin[(size_t)mm * COUT + c] = v + dn;
        } else {
          Ynext[(size_t)mm * COUT + c] = f2bf(v);
        }
      }
    }
  }
}

// ---------------------------------------------------------------------------
// Launch
// ---------------------------------------------------------------------------
extern "C" void kernel_launch(void* const* d_in, const int* in_sizes, int n_in,
                              void* d_out, int out_size, void* d_ws, size_t ws_size,
                              hipStream_t stream) {
  const float* feats = (const float*)d_in[0];
  const float* W0    = (const float*)d_in[1];   // [27,128,128]
  const float* W1    = (const float*)d_in[2];   // [27,128,32]
  const float* W2    = (const float*)d_in[3];   // [27,32,32]
  const int*   nbr   = (const int*)d_in[4];     // [27,N]
  float* out = (float*)d_out;                   // [N,32] fp32

  char* ws = (char*)d_ws;
  size_t off = 0;
  auto alloc = [&](size_t bytes) -> char* {
    char* p = ws + off;
    off += (bytes + 255) & ~(size_t)255;
    return p;
  };
  bf16_t* X0  = (bf16_t*)alloc((size_t)(N_VOX + 1) * 128 * 2);  // 76.8 MB
  bf16_t* X1  = (bf16_t*)alloc((size_t)(N_VOX + 1) * 128 * 2);  // 76.8 MB
  bf16_t* X2  = (bf16_t*)alloc((size_t)(N_VOX + 1) * 32 * 2);   // 19.2 MB
  bf16_t* WR0 = (bf16_t*)alloc((size_t)GK * 4 * 8 * 512 * 2);   // 884 KB
  bf16_t* WR1 = (bf16_t*)alloc((size_t)GK * 4 * 2 * 512 * 2);   // 221 KB
  bf16_t* WR2 = (bf16_t*)alloc((size_t)GK * 1 * 2 * 512 * 2);   // 55 KB

  // Weight repack (fp32 -> bf16 WMMA fragment order)
  {
    int t0 = GK * 4 * 8 * 32, t1 = GK * 4 * 2 * 32, t2 = GK * 1 * 2 * 32;
    repack_kernel<<<(t0 + 255) / 256, 256, 0, stream>>>(W0, WR0, 128, 128);
    repack_kernel<<<(t1 + 255) / 256, 256, 0, stream>>>(W1, WR1, 128, 32);
    repack_kernel<<<(t2 + 255) / 256, 256, 0, stream>>>(W2, WR2, 32, 32);
  }

  // Downsample + residual -> X0 (and zero rows N of X0/X1/X2)
  {
    int waves = N_VOX + 1;
    prep_kernel<<<(waves + 7) / 8, 256, 0, stream>>>(feats, X0, X1, X2);
  }

  // conv0: MT=1 (8 tiles/block); conv1/conv2: MT=2 (16 tiles/block)
  const int blocks1 = (TILES + 7) / 8;    // 2344
  const int blocks2 = (TILES + 15) / 16;  // 1172
  conv_kernel<128, 128, 1, false><<<blocks1, 256, 0, stream>>>(
      X0, WR0, nbr, X1, nullptr, nullptr);
  conv_kernel<128, 32, 2, false><<<blocks2, 256, 0, stream>>>(
      X1, WR1, nbr, X2, nullptr, nullptr);
  conv_kernel<32, 32, 2, true><<<blocks2, 256, 0, stream>>>(
      X2, WR2, nbr, nullptr, out, feats);
}
